// Cov2GEN_24601572671981
// MI455X (gfx1250) — matrix operand order repaired and checked by reading, not compiled, not run on hardware
//
#include <hip/hip_runtime.h>

// ---------------------------------------------------------------------------
// Problem constants (match reference)
// ---------------------------------------------------------------------------
#define NN    20000
#define EE    320000
#define HDIM  256
#define GG    128
#define FNODE 64
#define FEDGE 16
#define OUTD  10

typedef __attribute__((ext_vector_type(16))) _Float16 v16h;
typedef __attribute__((ext_vector_type(8)))  _Float16 v8h;
typedef __attribute__((ext_vector_type(4)))  _Float16 v4h;
typedef __attribute__((ext_vector_type(8)))  float    v8f;

// ---------------------------------------------------------------------------
// Monotonic uint encoding of float for order-independent atomic max
// ---------------------------------------------------------------------------
__device__ __forceinline__ unsigned fkey(float f) {
  unsigned u = __float_as_uint(f);
  return (u & 0x80000000u) ? ~u : (u | 0x80000000u);
}
__device__ __forceinline__ float fdec(unsigned k) {
  unsigned u = (k & 0x80000000u) ? (k & 0x7fffffffu) : ~k;
  return __uint_as_float(u);
}
// fkey(-FLT_MAX) == 0x00800000u  (used as "-inf" init)
#define NEG_MAX_KEY 0x00800000u

// ---------------------------------------------------------------------------
// Small utility kernels
// ---------------------------------------------------------------------------
__global__ void fill_u32_k(unsigned* __restrict__ p, unsigned v, long n) {
  long i = (long)blockIdx.x * blockDim.x + threadIdx.x;
  if (i < n) p[i] = v;
}

// f32 [rows,Kin] -> f16 [rows,Kout] zero-padded (Kout power of two, koutLog)
__global__ void cvt_pad_f16_k(const float* __restrict__ s, _Float16* __restrict__ d,
                              int Kin, int koutLog, long total) {
  long i = (long)blockIdx.x * blockDim.x + threadIdx.x;
  if (i >= total) return;
  int k = (int)(i & ((1 << koutLog) - 1));
  long row = i >> koutLog;
  d[i] = (k < Kin) ? (_Float16)s[row * Kin + k] : (_Float16)0.0f;
}

// ---------------------------------------------------------------------------
// Wave-level WMMA GEMM, no LDS, no bounds checks (K % 32 == 0 by construction).
//   OUT = act( A[M,K] @ W[*,K]^T + bias )
// Fragment k order used here (contiguous, k = kc+16*half+s, same for A and B):
//   A frag: lane L (row M=L%16, half=L/16): runs [kc+half*16,+8) & [kc+half*16+8,+8)
//   B frag: lane L (col N=L%16, kh=L/16): run  [kc+kh*16,+16)  (one 32B load)
// Each wave computes NT adjacent 16x16 tiles (A frag reused NT times).
// MODE 0: LSTM gates, float out   y = sigmoid(z_i)*tanh(z_g)   (two B rows)
// MODE 1: LSTM gates, f16 out
// MODE 2: BatchNorm+ReLU out      y = relu((z+b)*gamma/sqrt(1+eps)+beta)
// MODE 3: bias+ReLU out           y = relu(z+b)
// ---------------------------------------------------------------------------
template <int MODE, int NT, int K, bool A16>
__global__ void gemm16_wmma_k(const void* __restrict__ Av, int M,
                              const _Float16* __restrict__ W,   // [rows, K] f16
                              const float* __restrict__ bias,
                              const float* __restrict__ gamma,
                              const float* __restrict__ beta,
                              float* __restrict__ outF,
                              _Float16* __restrict__ outH,
                              int ldo, int gateOff) {
  static_assert(K % 32 == 0, "K must be a multiple of 32");
  const int lane = threadIdx.x;                 // 0..31 (wave32)
  const int wid  = threadIdx.y;                 // 4 waves / block
  const int r0   = (blockIdx.x * 4 + wid) * 16; // M tile
  const int c0   = blockIdx.y * (16 * NT);      // N tile group
  if (r0 >= M) return;

  const int m    = lane & 15;
  const int half = lane >> 4;

  v8f acc[NT];
  v8f acc2[NT];
  #pragma unroll
  for (int t = 0; t < NT; ++t) {
    acc[t] = (v8f){};
    if (MODE <= 1) acc2[t] = (v8f){};
  }

  const float*    aRowF = (const float*)Av    + (long)(r0 + m) * K;
  const _Float16* aRowH = (const _Float16*)Av + (long)(r0 + m) * K;

  for (int kc = 0; kc < K; kc += 32) {
    // ---- A fragment: two contiguous 8-element runs ----
    v16h av;
    #pragma unroll
    for (int sgm = 0; sgm < 2; ++sgm) {
      const int kb = kc + (half << 4) + sgm * 8;    // < K (half<=1, K%32==0)
      if (A16) {
        const v8h seg = *(const v8h*)(aRowH + kb);
        #pragma unroll
        for (int j = 0; j < 8; ++j) av[sgm * 8 + j] = seg[j];
      } else {
        const float4 f0 = *(const float4*)(aRowF + kb);
        const float4 f1 = *(const float4*)(aRowF + kb + 4);
        av[sgm * 8 + 0] = (_Float16)f0.x;
        av[sgm * 8 + 1] = (_Float16)f0.y;
        av[sgm * 8 + 2] = (_Float16)f0.z;
        av[sgm * 8 + 3] = (_Float16)f0.w;
        av[sgm * 8 + 4] = (_Float16)f1.x;
        av[sgm * 8 + 5] = (_Float16)f1.y;
        av[sgm * 8 + 6] = (_Float16)f1.z;
        av[sgm * 8 + 7] = (_Float16)f1.w;
      }
    }

    // ---- B fragments (one 32B contiguous load each) + WMMA ----
    const int kb = kc + (half << 4);                // < K
    #pragma unroll
    for (int t = 0; t < NT; ++t) {
      const int col = c0 + t * 16 + m;              // this lane's output column
      const v16h bv = *(const v16h*)(W + (long)col * K + kb);
      acc[t] = __builtin_amdgcn_wmma_f32_16x16x32_f16(false, av, false, bv,
                                                      (short)0, acc[t], false, false);
      if (MODE <= 1) {
        const v16h bv2 = *(const v16h*)(W + (long)(gateOff + col) * K + kb);
        acc2[t] = __builtin_amdgcn_wmma_f32_16x16x32_f16(false, av, false, bv2,
                                                         (short)0, acc2[t], false, false);
      }
    }
  }

  // ---- epilogue: C/D layout lane N=L%16, vgpr v -> row M = v + 8*(L/16) ----
  #pragma unroll
  for (int t = 0; t < NT; ++t) {
    const int col = c0 + t * 16 + m;
    #pragma unroll
    for (int vg = 0; vg < 8; ++vg) {
      const int row = r0 + vg + (half << 3);
      float z = acc[t][vg];
      float y;
      if (MODE <= 1) {
        float zi = z + bias[col];
        float zg = acc2[t][vg] + bias[gateOff + col];
        y = (1.0f / (1.0f + __expf(-zi))) * tanhf(zg);
      } else if (MODE == 2) {
        float s = gamma[col] * rsqrtf(1.0f + 1e-5f);
        y = fmaxf((z + bias[col]) * s + beta[col], 0.0f);
      } else {
        y = fmaxf(z + bias[col], 0.0f);
      }
      if (MODE == 1) outH[(long)row * ldo + col] = (_Float16)y;
      else           outF[(long)row * ldo + col] = y;
    }
  }
}

// ---------------------------------------------------------------------------
// Segment-softmax message passing (GENConv aggr='softmax')
// Pass 1: per-(dst,channel) max of t*msg, msg = relu(h[src]+ea) + 1e-7
// ---------------------------------------------------------------------------
__global__ void msg_max_k(const float* __restrict__ h,
                          const _Float16* __restrict__ ea,
                          const int* __restrict__ src, const int* __restrict__ dst,
                          const float* __restrict__ tptr,
                          unsigned* __restrict__ mkey, long total) {
  long idx = (long)blockIdx.x * blockDim.x + threadIdx.x;
  if (idx >= total) return;
  int e = (int)(idx >> 6);
  int c = (int)(idx & 63) << 2;       // 4 channels per thread, H=256
  int s = src[e], d = dst[e];
  float t = tptr[0];
  const _Float16* ep = ea + (long)e * HDIM + c;
  __builtin_prefetch(ep + 16 * HDIM, 0, 1);     // global_prefetch_b8 on the stream
  const v4h ev = *(const v4h*)ep;
  const float4 hv = *(const float4*)(h + (long)s * HDIM + c);
  float msg0 = fmaxf(hv.x + (float)ev[0], 0.0f) + 1e-7f;
  float msg1 = fmaxf(hv.y + (float)ev[1], 0.0f) + 1e-7f;
  float msg2 = fmaxf(hv.z + (float)ev[2], 0.0f) + 1e-7f;
  float msg3 = fmaxf(hv.w + (float)ev[3], 0.0f) + 1e-7f;
  unsigned* mp = mkey + (long)d * HDIM + c;
  atomicMax(mp + 0, fkey(t * msg0));
  atomicMax(mp + 1, fkey(t * msg1));
  atomicMax(mp + 2, fkey(t * msg2));
  atomicMax(mp + 3, fkey(t * msg3));
}

// Pass 2: e = exp(t*msg - m[dst]); accumulate sum(e) and sum(e*msg)
__global__ void msg_sum_k(const float* __restrict__ h,
                          const _Float16* __restrict__ ea,
                          const int* __restrict__ src, const int* __restrict__ dst,
                          const float* __restrict__ tptr,
                          const unsigned* __restrict__ mkey,
                          float* __restrict__ ssum, float* __restrict__ wsum,
                          long total) {
  long idx = (long)blockIdx.x * blockDim.x + threadIdx.x;
  if (idx >= total) return;
  int e = (int)(idx >> 6);
  int c = (int)(idx & 63) << 2;
  int s = src[e], d = dst[e];
  float t = tptr[0];
  const _Float16* ep = ea + (long)e * HDIM + c;
  __builtin_prefetch(ep + 16 * HDIM, 0, 1);
  const v4h ev = *(const v4h*)ep;
  const float4 hv = *(const float4*)(h + (long)s * HDIM + c);
  const unsigned* mp = mkey + (long)d * HDIM + c;
  float* sp = ssum + (long)d * HDIM + c;
  float* wp = wsum + (long)d * HDIM + c;
  float hvv[4] = {hv.x, hv.y, hv.z, hv.w};
  #pragma unroll
  for (int i = 0; i < 4; ++i) {
    float msg = fmaxf(hvv[i] + (float)ev[i], 0.0f) + 1e-7f;
    float ex = __expf(t * msg - fdec(mp[i]));
    atomicAdd(sp + i, ex);
    atomicAdd(wp + i, ex * msg);
  }
}

// h_comb = h + aggr   (aggr = wsum / (ssum + 1e-16))
__global__ void combine_k(const float* __restrict__ h,
                          const float* __restrict__ wsum,
                          const float* __restrict__ ssum,
                          float* __restrict__ hcomb, long n) {
  long i = (long)blockIdx.x * blockDim.x + threadIdx.x;
  if (i < n) hcomb[i] = h[i] + wsum[i] / (ssum[i] + 1e-16f);
}

// ---------------------------------------------------------------------------
// Instance norm per (graph, channel)
// ---------------------------------------------------------------------------
__global__ void count_k(const int* __restrict__ batch, int* __restrict__ gcnt, int n) {
  int i = blockIdx.x * blockDim.x + threadIdx.x;
  if (i < n) atomicAdd(&gcnt[batch[i]], 1);
}

__global__ void inorm_acc_k(const float* __restrict__ x, const int* __restrict__ batch,
                            float* __restrict__ gsum, float* __restrict__ gsq, long n) {
  long i = (long)blockIdx.x * blockDim.x + threadIdx.x;
  if (i >= n) return;
  int node = (int)(i >> 8);   // H = 256
  int c = (int)(i & 255);
  int b = batch[node];
  float v = x[i];
  atomicAdd(&gsum[(long)b * HDIM + c], v);
  atomicAdd(&gsq[(long)b * HDIM + c], v * v);
}

__global__ void inorm_norm_k(float* __restrict__ x, const int* __restrict__ batch,
                             const float* __restrict__ gsum, const float* __restrict__ gsq,
                             const int* __restrict__ gcnt, long n) {
  long i = (long)blockIdx.x * blockDim.x + threadIdx.x;
  if (i >= n) return;
  int node = (int)(i >> 8);
  int c = (int)(i & 255);
  int b = batch[node];
  float cnt = fmaxf((float)gcnt[b], 1.0f);
  float mean = gsum[(long)b * HDIM + c] / cnt;
  float var = gsq[(long)b * HDIM + c] / cnt - mean * mean;
  x[i] = (x[i] - mean) * rsqrtf(var + 1e-5f);
}

// ---------------------------------------------------------------------------
// Global max pool per graph + output linear + sigmoid
// ---------------------------------------------------------------------------
__global__ void pool_k(const float* __restrict__ x, const int* __restrict__ batch,
                       unsigned* __restrict__ pkey, long n) {
  long i = (long)blockIdx.x * blockDim.x + threadIdx.x;
  if (i >= n) return;
  int node = (int)(i >> 8);
  int c = (int)(i & 255);
  atomicMax(&pkey[(long)batch[node] * HDIM + c], fkey(x[i]));
}

__global__ void final_k(const unsigned* __restrict__ pkey,
                        const float* __restrict__ lw, const float* __restrict__ lb,
                        float* __restrict__ out, int total) {
  int i = blockIdx.x * blockDim.x + threadIdx.x;
  if (i >= total) return;
  int g = i / OUTD, o = i - g * OUTD;
  float acc = lb[o];
  for (int c = 0; c < HDIM; ++c) {
    float p = fdec(pkey[(long)g * HDIM + c]);
    p = (p < -1e37f) ? 0.0f : p;           // empty segment -> 0 (isfinite mask)
    acc += p * lw[o * HDIM + c];
  }
  out[i] = 1.0f / (1.0f + __expf(-acc));
}

// ---------------------------------------------------------------------------
// Host-side orchestration
// ---------------------------------------------------------------------------
extern "C" void kernel_launch(void* const* d_in, const int* in_sizes, int n_in,
                              void* d_out, int out_size, void* d_ws, size_t ws_size,
                              hipStream_t stream) {
  const float* x         = (const float*)d_in[0];
  const int*   eidx      = (const int*)d_in[1];
  const float* edge_attr = (const float*)d_in[2];
  const int*   batch     = (const int*)d_in[3];
  const float* node_W    = (const float*)d_in[4];
  const float* node_b    = (const float*)d_in[5];
  const float* edge_W    = (const float*)d_in[6];
  const float* edge_b    = (const float*)d_in[7];
  const float* t1        = (const float*)d_in[8];
  const float* g1_w1     = (const float*)d_in[9];
  const float* g1_b1     = (const float*)d_in[10];
  const float* g1_gamma  = (const float*)d_in[11];
  const float* g1_beta   = (const float*)d_in[12];
  const float* g1_w2     = (const float*)d_in[13];
  const float* g1_b2     = (const float*)d_in[14];
  const float* t2        = (const float*)d_in[15];
  const float* g2_w1     = (const float*)d_in[16];
  const float* g2_b1     = (const float*)d_in[17];
  const float* g2_gamma  = (const float*)d_in[18];
  const float* g2_beta   = (const float*)d_in[19];
  const float* g2_w2     = (const float*)d_in[20];
  const float* g2_b2     = (const float*)d_in[21];
  const float* lin_w     = (const float*)d_in[22];
  const float* lin_b     = (const float*)d_in[23];
  float* out = (float*)d_out;

  const int* src  = eidx;
  const int* dstI = eidx + EE;

  // -------- carve workspace --------
  char* base = (char*)d_ws;
  size_t off = 0;
  auto carve = [&](size_t bytes) -> void* {
    void* p = base + off;
    off += (bytes + 255) & ~(size_t)255;
    return p;
  };
  _Float16* ea16   = (_Float16*)carve((size_t)EE * HDIM * 2); // edge embedding (f16)
  _Float16* eat16  = (_Float16*)carve((size_t)EE * 32 * 2);   // edge_attr f16, K padded 16->32
  float*    h      = (float*)carve((size_t)NN * HDIM * 4);    // node features
  float*    hcomb  = (float*)carve((size_t)NN * HDIM * 4);    // h + aggr
  float*    h2     = (float*)carve((size_t)NN * 2 * HDIM * 4);// MLP hidden
  unsigned* mkey   = (unsigned*)carve((size_t)NN * HDIM * 4); // softmax max keys
  float*    ssum   = (float*)carve((size_t)NN * HDIM * 4);    // sum exp
  float*    wsum   = (float*)carve((size_t)NN * HDIM * 4);    // sum exp*msg
  float*    gsum   = (float*)carve((size_t)GG * HDIM * 4);
  float*    gsq    = (float*)carve((size_t)GG * HDIM * 4);
  int*      gcnt   = (int*)carve((size_t)GG * 4);
  unsigned* pkey   = (unsigned*)carve((size_t)GG * HDIM * 4);
  _Float16* nodeW16 = (_Float16*)carve((size_t)4 * HDIM * FNODE * 2);
  _Float16* edgeW16 = (_Float16*)carve((size_t)4 * HDIM * 32 * 2);  // K padded 16->32
  _Float16* w11_16  = (_Float16*)carve((size_t)2 * HDIM * HDIM * 2);
  _Float16* w12_16  = (_Float16*)carve((size_t)HDIM * 2 * HDIM * 2);
  _Float16* w21_16  = (_Float16*)carve((size_t)2 * HDIM * HDIM * 2);
  _Float16* w22_16  = (_Float16*)carve((size_t)HDIM * 2 * HDIM * 2);

  const long NH = (long)NN * HDIM;
  const long totalE = (long)EE * (HDIM / 4);
  auto blks = [](long n) { return (int)((n + 255) / 256); };
  auto mtiles = [](int mrows) { return (mrows / 16 + 3) / 4; }; // 4 waves/block

  // -------- weights / edge_attr -> f16 (zero-padded where K%32 != 0) --------
  cvt_pad_f16_k<<<blks((long)EE * 32), 256, 0, stream>>>(edge_attr, eat16, FEDGE, 5, (long)EE * 32);
  cvt_pad_f16_k<<<blks(4 * HDIM * FNODE), 256, 0, stream>>>(node_W, nodeW16, FNODE, 6, 4 * HDIM * FNODE);
  cvt_pad_f16_k<<<blks(4 * HDIM * 32), 256, 0, stream>>>(edge_W, edgeW16, FEDGE, 5, 4 * HDIM * 32);
  cvt_pad_f16_k<<<blks(2 * HDIM * HDIM), 256, 0, stream>>>(g1_w1, w11_16, HDIM, 8, 2 * HDIM * HDIM);
  cvt_pad_f16_k<<<blks(2 * HDIM * HDIM), 256, 0, stream>>>(g1_w2, w12_16, 2 * HDIM, 9, 2 * HDIM * HDIM);
  cvt_pad_f16_k<<<blks(2 * HDIM * HDIM), 256, 0, stream>>>(g2_w1, w21_16, HDIM, 8, 2 * HDIM * HDIM);
  cvt_pad_f16_k<<<blks(2 * HDIM * HDIM), 256, 0, stream>>>(g2_w2, w22_16, 2 * HDIM, 9, 2 * HDIM * HDIM);

  const dim3 wblk(32, 4);

  // -------- LSTM-cell embeddings (WMMA): h = sigmoid(z_i)*tanh(z_g) --------
  gemm16_wmma_k<0, 2, FNODE, false><<<dim3(mtiles(NN), HDIM / 32), wblk, 0, stream>>>(
      x, NN, nodeW16, node_b, nullptr, nullptr, h, nullptr, HDIM, 2 * HDIM);
  gemm16_wmma_k<1, 2, 32, true><<<dim3(mtiles(EE), HDIM / 32), wblk, 0, stream>>>(
      eat16, EE, edgeW16, edge_b, nullptr, nullptr, nullptr, ea16, HDIM, 2 * HDIM);

  // -------- GENConv layer --------
  auto run_conv = [&](const float* tptr, const _Float16* W1, const float* b1,
                      const float* gamma, const float* beta,
                      const _Float16* W2, const float* b2) {
    fill_u32_k<<<blks(NH), 256, 0, stream>>>(mkey, NEG_MAX_KEY, NH);
    msg_max_k<<<blks(totalE), 256, 0, stream>>>(h, ea16, src, dstI, tptr, mkey, totalE);
    fill_u32_k<<<blks(NH), 256, 0, stream>>>((unsigned*)ssum, 0u, NH);
    fill_u32_k<<<blks(NH), 256, 0, stream>>>((unsigned*)wsum, 0u, NH);
    msg_sum_k<<<blks(totalE), 256, 0, stream>>>(h, ea16, src, dstI, tptr, mkey,
                                                ssum, wsum, totalE);
    combine_k<<<blks(NH), 256, 0, stream>>>(h, wsum, ssum, hcomb, NH);
    gemm16_wmma_k<2, 4, HDIM, false><<<dim3(mtiles(NN), 2 * HDIM / 64), wblk, 0, stream>>>(
        hcomb, NN, W1, b1, gamma, beta, h2, nullptr, 2 * HDIM, 0);
    gemm16_wmma_k<3, 4, 2 * HDIM, false><<<dim3(mtiles(NN), HDIM / 64), wblk, 0, stream>>>(
        h2, NN, W2, b2, nullptr, nullptr, h, nullptr, HDIM, 0);
  };

  auto run_inorm = [&]() {
    fill_u32_k<<<blks(GG * HDIM), 256, 0, stream>>>((unsigned*)gsum, 0u, GG * HDIM);
    fill_u32_k<<<blks(GG * HDIM), 256, 0, stream>>>((unsigned*)gsq, 0u, GG * HDIM);
    fill_u32_k<<<1, 256, 0, stream>>>((unsigned*)gcnt, 0u, GG);
    count_k<<<blks(NN), 256, 0, stream>>>(batch, gcnt, NN);
    inorm_acc_k<<<blks(NH), 256, 0, stream>>>(h, batch, gsum, gsq, NH);
    inorm_norm_k<<<blks(NH), 256, 0, stream>>>(h, batch, gsum, gsq, gcnt, NH);
  };

  run_conv(t1, w11_16, g1_b1, g1_gamma, g1_beta, w12_16, g1_b2);
  run_inorm();
  run_conv(t2, w21_16, g2_b1, g2_gamma, g2_beta, w22_16, g2_b2);
  run_inorm();

  // -------- pool + output linear + sigmoid --------
  fill_u32_k<<<blks(GG * HDIM), 256, 0, stream>>>(pkey, NEG_MAX_KEY, GG * HDIM);
  pool_k<<<blks(NH), 256, 0, stream>>>(h, batch, pkey, NH);
  final_k<<<blks(GG * OUTD), 256, 0, stream>>>(pkey, lin_w, lin_b, out, GG * OUTD);
}
